// GraphNet_1_38826504355978
// MI455X (gfx1250) — matrix operand order repaired and checked by reading, not compiled, not run on hardware
//
#include <hip/hip_runtime.h>

#define N_NODES 50000
#define N_EDGES 800000
#define F_IN 64
#define HD 64
#define N_GRAPHS 64
#define BN_EPS 1e-5f
#define ENC_NEG_INF 0x007FFFFFu

typedef __attribute__((ext_vector_type(16))) __bf16 bf16x16;
typedef __attribute__((ext_vector_type(8)))  __bf16 bf16x8;
typedef __attribute__((ext_vector_type(8)))  float  floatx8;

// ---------- helpers ----------
__device__ __forceinline__ unsigned encf(float f) {
    unsigned b = __float_as_uint(f);
    return (b & 0x80000000u) ? ~b : (b | 0x80000000u);
}
__device__ __forceinline__ float decf(unsigned u) {
    unsigned b = (u & 0x80000000u) ? (u & 0x7FFFFFFFu) : ~u;
    return __uint_as_float(b);
}

// A fragment (16-bit A, 16x32): lane l: row l&15; low lanes K={0..7,16..23}, high lanes K={8..15,24..31}
__device__ __forceinline__ bf16x16 frag_a_load(const __bf16* base, int ld, int lane, int kstep) {
    int r  = lane & 15;
    int kb = kstep * 32 + ((lane & 16) ? 8 : 0);
    const bf16x8* p = (const bf16x8*)(base + r * ld + kb);
    bf16x8 lo = p[0];
    bf16x8 hi = p[2];  // +16 elements
    bf16x16 f;
#pragma unroll
    for (int i = 0; i < 8; ++i) { f[i] = lo[i]; f[i + 8] = hi[i]; }
    return f;
}

// B fragment (16-bit B, 32x16): lane l: col l&15; low lanes K={0..7,16..23}, high lanes K={8..15,24..31}
// wt is [N][K] row-major (pre-transposed weights)
__device__ __forceinline__ bf16x16 frag_b_load(const __bf16* wt, int ld, int lane, int kstep, int colbase) {
    int c  = colbase + (lane & 15);
    int kb = kstep * 32 + ((lane & 16) ? 8 : 0);
    const bf16x8* p = (const bf16x8*)(wt + c * ld + kb);
    bf16x8 lo = p[0];
    bf16x8 hi = p[2];
    bf16x16 f;
#pragma unroll
    for (int i = 0; i < 8; ++i) { f[i] = lo[i]; f[i + 8] = hi[i]; }
    return f;
}

// ---------- small utility kernels ----------
__global__ void k_fill_u32(unsigned* p, unsigned v, int n) {
    int i = blockIdx.x * blockDim.x + threadIdx.x;
    if (i < n) p[i] = v;
}
__global__ void k_fill_f32(float* p, float v, int n) {
    int i = blockIdx.x * blockDim.x + threadIdx.x;
    if (i < n) p[i] = v;
}
// transpose (K,N) f32 -> (N,K) bf16
__global__ void k_prepw(const float* w, __bf16* wt, int K, int N) {
    int i = blockIdx.x * blockDim.x + threadIdx.x;
    if (i < K * N) {
        int k = i / N, n = i % N;
        wt[n * K + k] = (__bf16)w[i];
    }
}
// fold BN batch stats into per-channel scale/shift
__global__ void k_stats(const float* sum, const float* sq, const float* gamma, const float* beta,
                        float* sc, float* sh, int n) {
    int c = threadIdx.x;
    if (c < n) {
        float m   = sum[c] * (1.0f / (float)N_EDGES);
        float var = sq[c] * (1.0f / (float)N_EDGES) - m * m;
        float s   = gamma[c] * rsqrtf(var + BN_EPS);
        sc[c] = s;
        sh[c] = beta[c] - m * s;
    }
}

// ---------- GEMM 1a: e=[x_i, x_j-x_i] (K=128) @ W1a (N=128), bias+ReLU, stats ----------
__global__ __launch_bounds__(256) void k_gemm1a(const float* __restrict__ x, const int* __restrict__ ei,
                                                const float* __restrict__ bias_g,
                                                const __bf16* __restrict__ wt,
                                                __bf16* __restrict__ hout,
                                                float* __restrict__ gsum, float* __restrict__ gsq) {
    __shared__ __bf16 sW[128 * 128];
    __shared__ __bf16 sA[16 * 128];
    __shared__ __bf16 sO[16 * 128];
    __shared__ float  lsum[128], lsq[128];

    int tid = threadIdx.x;
    for (int i = tid; i < 128 * 128 / 8; i += 256)
        *(bf16x8*)&sW[i * 8] = *(const bf16x8*)&wt[i * 8];
    if (tid < 128) { lsum[tid] = 0.f; lsq[tid] = 0.f; }
    __syncthreads();

    int lane = tid & 31, wv = tid >> 5;  // 8 waves: one 16-col tile each
    const int ntiles = N_EDGES / 16;
    for (int t = blockIdx.x; t < ntiles; t += gridDim.x) {
        int e0 = t * 16;
        {   // build A tile: 16 threads/edge, 8 channels each
            int e   = e0 + (tid >> 4);
            int src = ei[e];
            int dst = ei[N_EDGES + e];
            int cb  = (tid & 15) * 8;
            const float* xd = x + (size_t)dst * F_IN;
            const float* xs = x + (size_t)src * F_IN;
            __bf16* q = &sA[(tid >> 4) * 128 + cb];
            if (cb < F_IN) {
#pragma unroll
                for (int i = 0; i < 8; ++i) q[i] = (__bf16)xd[cb + i];
            } else {
                int c = cb - F_IN;
#pragma unroll
                for (int i = 0; i < 8; ++i) q[i] = (__bf16)(xs[c + i] - xd[c + i]);
            }
        }
        __syncthreads();

        floatx8 acc = {0.f, 0.f, 0.f, 0.f, 0.f, 0.f, 0.f, 0.f};
#pragma unroll
        for (int ks = 0; ks < 4; ++ks) {
            bf16x16 a = frag_a_load(sA, 128, lane, ks);
            bf16x16 b = frag_b_load(sW, 128, lane, ks, wv * 16);
            acc = __builtin_amdgcn_wmma_f32_16x16x32_bf16(false, a, false, b, (short)0, acc, false, false);
        }
        // epilogue: bias + relu + stats + stage to LDS
        int   n    = wv * 16 + (lane & 15);
        float bval = bias_g[n];
        int   mb   = (lane & 16) ? 8 : 0;
        float s = 0.f, s2 = 0.f;
#pragma unroll
        for (int i = 0; i < 8; ++i) {
            float v = acc[i] + bval;
            v = v > 0.f ? v : 0.f;
            s += v; s2 += v * v;
            sO[(mb + i) * 128 + n] = (__bf16)v;
        }
        s  += __shfl_xor(s, 16);
        s2 += __shfl_xor(s2, 16);
        if (lane < 16) { atomicAdd(&lsum[n], s); atomicAdd(&lsq[n], s2); }
        __syncthreads();
        // coalesced store: 16x128 bf16 = 256 x 16B
        *(bf16x8*)&hout[(size_t)e0 * 128 + tid * 8] = *(bf16x8*)&sO[tid * 8];
        __syncthreads();
    }
    if (tid < 128) { atomicAdd(&gsum[tid], lsum[tid]); atomicAdd(&gsq[tid], lsq[tid]); }
}

// ---------- GEMM 1b: BN(h1a) (K=128) @ W1b (N=64), bias+ReLU, stats ----------
__global__ __launch_bounds__(256) void k_gemm1b(const __bf16* __restrict__ hin,
                                                const float* __restrict__ sc, const float* __restrict__ sh,
                                                const float* __restrict__ bias_g,
                                                const __bf16* __restrict__ wt,
                                                __bf16* __restrict__ hout,
                                                float* __restrict__ gsum, float* __restrict__ gsq) {
    __shared__ __bf16 sW[64 * 128];
    __shared__ __bf16 sA[32 * 128];
    __shared__ __bf16 sO[32 * 64];
    __shared__ float  lsum[64], lsq[64], ssc[128], ssh[128];

    int tid = threadIdx.x;
    for (int i = tid; i < 64 * 128 / 8; i += 256)
        *(bf16x8*)&sW[i * 8] = *(const bf16x8*)&wt[i * 8];
    if (tid < 128) { ssc[tid] = sc[tid]; ssh[tid] = sh[tid]; }
    if (tid < 64)  { lsum[tid] = 0.f; lsq[tid] = 0.f; }
    __syncthreads();

    int lane = tid & 31, wv = tid >> 5;
    int rt = wv >> 2, ct = wv & 3;  // 2 row tiles x 4 col tiles
    const int ntiles = N_EDGES / 32;
    for (int t = blockIdx.x; t < ntiles; t += gridDim.x) {
        size_t e0 = (size_t)t * 32;
        {   // build A with BN affine applied
            int r  = tid >> 3;
            int cb = (tid & 7) * 16;
            const __bf16* p = &hin[(e0 + r) * 128 + cb];
            __bf16* q = &sA[r * 128 + cb];
#pragma unroll
            for (int i = 0; i < 16; ++i)
                q[i] = (__bf16)((float)p[i] * ssc[cb + i] + ssh[cb + i]);
        }
        __syncthreads();

        floatx8 acc = {0.f, 0.f, 0.f, 0.f, 0.f, 0.f, 0.f, 0.f};
#pragma unroll
        for (int ks = 0; ks < 4; ++ks) {
            bf16x16 a = frag_a_load(&sA[rt * 16 * 128], 128, lane, ks);
            bf16x16 b = frag_b_load(sW, 128, lane, ks, ct * 16);
            acc = __builtin_amdgcn_wmma_f32_16x16x32_bf16(false, a, false, b, (short)0, acc, false, false);
        }
        int   n    = ct * 16 + (lane & 15);
        float bval = bias_g[n];
        int   mb   = rt * 16 + ((lane & 16) ? 8 : 0);
        float s = 0.f, s2 = 0.f;
#pragma unroll
        for (int i = 0; i < 8; ++i) {
            float v = acc[i] + bval;
            v = v > 0.f ? v : 0.f;
            s += v; s2 += v * v;
            sO[(mb + i) * 64 + n] = (__bf16)v;
        }
        s  += __shfl_xor(s, 16);
        s2 += __shfl_xor(s2, 16);
        if (lane < 16) { atomicAdd(&lsum[n], s); atomicAdd(&lsq[n], s2); }
        __syncthreads();
        *(bf16x8*)&hout[e0 * 64 + tid * 8] = *(bf16x8*)&sO[tid * 8];
        __syncthreads();
    }
    if (tid < 64) { atomicAdd(&gsum[tid], lsum[tid]); atomicAdd(&gsq[tid], lsq[tid]); }
}

// ---------- GEMM 2: e=[x1_i, x1_j-x1_i] (K=128) @ W2 (N=64), bias+ReLU, stats ----------
__global__ __launch_bounds__(256) void k_gemm2(const float* __restrict__ x1, const int* __restrict__ ei,
                                               const float* __restrict__ bias_g,
                                               const __bf16* __restrict__ wt,
                                               __bf16* __restrict__ hout,
                                               float* __restrict__ gsum, float* __restrict__ gsq) {
    __shared__ __bf16 sW[64 * 128];
    __shared__ __bf16 sA[32 * 128];
    __shared__ __bf16 sO[32 * 64];
    __shared__ float  lsum[64], lsq[64];

    int tid = threadIdx.x;
    for (int i = tid; i < 64 * 128 / 8; i += 256)
        *(bf16x8*)&sW[i * 8] = *(const bf16x8*)&wt[i * 8];
    if (tid < 64) { lsum[tid] = 0.f; lsq[tid] = 0.f; }
    __syncthreads();

    int lane = tid & 31, wv = tid >> 5;
    int rt = wv >> 2, ct = wv & 3;
    const int ntiles = N_EDGES / 32;
    for (int t = blockIdx.x; t < ntiles; t += gridDim.x) {
        size_t e0 = (size_t)t * 32;
        {   // build A tile by gather
            int r   = tid >> 3;
            int e   = (int)e0 + r;
            int src = ei[e];
            int dst = ei[N_EDGES + e];
            int cb  = (tid & 7) * 16;
            const float* xd = x1 + (size_t)dst * HD;
            const float* xs = x1 + (size_t)src * HD;
            __bf16* q = &sA[r * 128 + cb];
            if (cb < HD) {
#pragma unroll
                for (int i = 0; i < 16; ++i) q[i] = (__bf16)xd[cb + i];
            } else {
                int c = cb - HD;
#pragma unroll
                for (int i = 0; i < 16; ++i) q[i] = (__bf16)(xs[c + i] - xd[c + i]);
            }
        }
        __syncthreads();

        floatx8 acc = {0.f, 0.f, 0.f, 0.f, 0.f, 0.f, 0.f, 0.f};
#pragma unroll
        for (int ks = 0; ks < 4; ++ks) {
            bf16x16 a = frag_a_load(&sA[rt * 16 * 128], 128, lane, ks);
            bf16x16 b = frag_b_load(sW, 128, lane, ks, ct * 16);
            acc = __builtin_amdgcn_wmma_f32_16x16x32_bf16(false, a, false, b, (short)0, acc, false, false);
        }
        int   n    = ct * 16 + (lane & 15);
        float bval = bias_g[n];
        int   mb   = rt * 16 + ((lane & 16) ? 8 : 0);
        float s = 0.f, s2 = 0.f;
#pragma unroll
        for (int i = 0; i < 8; ++i) {
            float v = acc[i] + bval;
            v = v > 0.f ? v : 0.f;
            s += v; s2 += v * v;
            sO[(mb + i) * 64 + n] = (__bf16)v;
        }
        s  += __shfl_xor(s, 16);
        s2 += __shfl_xor(s2, 16);
        if (lane < 16) { atomicAdd(&lsum[n], s); atomicAdd(&lsq[n], s2); }
        __syncthreads();
        *(bf16x8*)&hout[e0 * 64 + tid * 8] = *(bf16x8*)&sO[tid * 8];
        __syncthreads();
    }
    if (tid < 64) { atomicAdd(&gsum[tid], lsum[tid]); atomicAdd(&gsq[tid], lsq[tid]); }
}

// ---------- scatter-max (BN applied on the fly), encoded uint atomicMax ----------
__global__ void k_scatter(const __bf16* __restrict__ h, const float* __restrict__ sc,
                          const float* __restrict__ sh, const int* __restrict__ dstidx,
                          unsigned* __restrict__ agg) {
    const int total = N_EDGES * HD;
    for (int i = blockIdx.x * blockDim.x + threadIdx.x; i < total; i += gridDim.x * blockDim.x) {
        int e = i >> 6, c = i & 63;
        float v = (float)h[i] * sc[c] + sh[c];
        atomicMax(&agg[(size_t)dstidx[e] * HD + c], encf(v));
    }
}

// decode agg -> relu(where(isfinite, agg, 0))
__global__ void k_decode(const unsigned* __restrict__ agg, float* __restrict__ xo) {
    int i = blockIdx.x * blockDim.x + threadIdx.x;
    if (i < N_NODES * HD) {
        float v = decf(agg[i]);
        if (!__builtin_isfinite(v)) v = 0.f;
        xo[i] = v > 0.f ? v : 0.f;
    }
}

// fuse x1+x2 and pool per graph
__global__ void k_pool(const float* __restrict__ x1, const unsigned* __restrict__ x2agg,
                       const int* __restrict__ batch, unsigned* __restrict__ gm,
                       float* __restrict__ gs, float* __restrict__ cnt) {
    int i = blockIdx.x * blockDim.x + threadIdx.x;
    if (i < N_NODES * HD) {
        int n = i >> 6, c = i & 63;
        float v2 = decf(x2agg[i]);
        if (!__builtin_isfinite(v2)) v2 = 0.f;
        v2 = v2 > 0.f ? v2 : 0.f;
        float f = x1[i] + v2;
        int g = batch[n];
        atomicMax(&gm[g * HD + c], encf(f));
        atomicAdd(&gs[g * HD + c], f);
        if (c == 0) atomicAdd(&cnt[g], 1.0f);
    }
}

__global__ void k_final(const unsigned* __restrict__ gm, const float* __restrict__ gs,
                        const float* __restrict__ cnt, float* __restrict__ out) {
    int i = blockIdx.x * blockDim.x + threadIdx.x;
    if (i < N_GRAPHS * 2 * HD) {
        int g = i >> 7, c = i & 127;
        if (c < HD) {
            float v = decf(gm[g * HD + c]);
            if (!__builtin_isfinite(v)) v = 0.f;
            out[i] = v;
        } else {
            out[i] = gs[g * HD + (c - HD)] / fmaxf(cnt[g], 1.0f);
        }
    }
}

// ---------- launcher ----------
extern "C" void kernel_launch(void* const* d_in, const int* in_sizes, int n_in,
                              void* d_out, int out_size, void* d_ws, size_t ws_size,
                              hipStream_t stream) {
    const float* x    = (const float*)d_in[0];
    const int*   ei   = (const int*)d_in[1];
    const int*   bat  = (const int*)d_in[2];
    const float* W1a  = (const float*)d_in[3];
    const float* b1a  = (const float*)d_in[4];
    const float* g1a  = (const float*)d_in[5];
    const float* bt1a = (const float*)d_in[6];
    const float* W1b  = (const float*)d_in[7];
    const float* b1b  = (const float*)d_in[8];
    const float* g1b  = (const float*)d_in[9];
    const float* bt1b = (const float*)d_in[10];
    const float* W2   = (const float*)d_in[11];
    const float* b2   = (const float*)d_in[12];
    const float* g2   = (const float*)d_in[13];
    const float* bt2  = (const float*)d_in[14];
    float* out = (float*)d_out;

    char* ws = (char*)d_ws;
    size_t off = 0;
    auto take = [&](size_t bytes) { char* p = ws + off; off += (bytes + 255) & ~(size_t)255; return p; };

    __bf16*   h1a   = (__bf16*)take((size_t)N_EDGES * 128 * 2);
    __bf16*   h1b   = (__bf16*)take((size_t)N_EDGES * 64 * 2);   // reused as h2
    __bf16*   w1at  = (__bf16*)take(128 * 128 * 2);
    __bf16*   w1bt  = (__bf16*)take(64 * 128 * 2);
    __bf16*   w2t   = (__bf16*)take(64 * 128 * 2);
    float*    stat  = (float*)take(512 * 4);   // [sum1a|sq1a|sum1b|sq1b|sum2|sq2]
    float*    cs    = (float*)take(512 * 4);   // [sc1a|sh1a|sc1b|sh1b|sc2|sh2]
    unsigned* x1agg = (unsigned*)take((size_t)N_NODES * HD * 4);
    unsigned* x2agg = (unsigned*)take((size_t)N_NODES * HD * 4);
    float*    x1    = (float*)take((size_t)N_NODES * HD * 4);
    unsigned* gm    = (unsigned*)take(N_GRAPHS * HD * 4);
    float*    gs    = (float*)take(N_GRAPHS * HD * 4);
    float*    cnt   = (float*)take(N_GRAPHS * 4);

    const int* dstidx = ei + N_EDGES;
    const int NV = N_NODES * HD;  // 3,200,000

    // ---- init ----
    k_fill_f32<<<2, 256, 0, stream>>>(stat, 0.f, 512);
    k_fill_u32<<<(NV + 255) / 256, 256, 0, stream>>>(x1agg, ENC_NEG_INF, NV);
    k_fill_u32<<<(NV + 255) / 256, 256, 0, stream>>>(x2agg, ENC_NEG_INF, NV);
    k_fill_u32<<<16, 256, 0, stream>>>(gm, ENC_NEG_INF, N_GRAPHS * HD);
    k_fill_f32<<<16, 256, 0, stream>>>(gs, 0.f, N_GRAPHS * HD);
    k_fill_f32<<<1, 64, 0, stream>>>(cnt, 0.f, N_GRAPHS);

    // ---- weight transpose + bf16 convert ----
    k_prepw<<<64, 256, 0, stream>>>(W1a, w1at, 128, 128);
    k_prepw<<<32, 256, 0, stream>>>(W1b, w1bt, 128, 64);
    k_prepw<<<32, 256, 0, stream>>>(W2, w2t, 128, 64);

    // ---- conv1 layer a ----
    k_gemm1a<<<1024, 256, 0, stream>>>(x, ei, b1a, w1at, h1a, stat + 0, stat + 128);
    k_stats<<<1, 128, 0, stream>>>(stat + 0, stat + 128, g1a, bt1a, cs + 0, cs + 128, 128);

    // ---- conv1 layer b ----
    k_gemm1b<<<1024, 256, 0, stream>>>(h1a, cs + 0, cs + 128, b1b, w1bt, h1b, stat + 256, stat + 320);
    k_stats<<<1, 64, 0, stream>>>(stat + 256, stat + 320, g1b, bt1b, cs + 256, cs + 320, 64);

    // ---- aggregate conv1 -> x1 ----
    k_scatter<<<4096, 256, 0, stream>>>(h1b, cs + 256, cs + 320, dstidx, x1agg);
    k_decode<<<(NV + 255) / 256, 256, 0, stream>>>(x1agg, x1);

    // ---- conv2 ----
    k_gemm2<<<1024, 256, 0, stream>>>(x1, ei, b2, w2t, h1b /*h2*/, stat + 384, stat + 448);
    k_stats<<<1, 64, 0, stream>>>(stat + 384, stat + 448, g2, bt2, cs + 384, cs + 448, 64);
    k_scatter<<<4096, 256, 0, stream>>>(h1b /*h2*/, cs + 384, cs + 448, dstidx, x2agg);

    // ---- fuse + pool + output ----
    k_pool<<<(NV + 255) / 256, 256, 0, stream>>>(x1, x2agg, bat, gm, gs, cnt);
    k_final<<<32, 256, 0, stream>>>(gm, gs, cnt, out);
}
